// DLSMNLayer_20590073217500
// MI455X (gfx1250) — compile-verified, matrix operand order checked by reading, not run on hardware
//
#include <hip/hip_runtime.h>

#ifndef USE_ASYNC
#define USE_ASYNC 1
#endif

// ---------------------------------------------------------------------------
// Types / WMMA helpers (CDNA5 / gfx1250, wave32)
// ---------------------------------------------------------------------------
typedef __bf16 bf16;
typedef __attribute__((ext_vector_type(16))) __bf16 v16bf;
typedef __attribute__((ext_vector_type(8)))  float  v8f;

__device__ inline v8f v8f_zero() {
  v8f z = {0.f, 0.f, 0.f, 0.f, 0.f, 0.f, 0.f, 0.f};
  return z;
}

__device__ inline bf16 f2bf(float f) {
  union { float f; unsigned u; } a; a.f = f;
  unsigned r = a.u + 0x7FFFu + ((a.u >> 16) & 1u);   // round-to-nearest-even
  unsigned short h = (unsigned short)(r >> 16);
  return __builtin_bit_cast(bf16, h);
}

struct alignas(32) Frag256 { uint4 a, b; };

// Load 16 contiguous bf16 (32B) -> one WMMA fragment register set.
// A layout used here (contiguous k): lane holds row (lane&15), K chunk (lane>>4)*16.
// B fragments use the identical pattern when the tile is staged [n][k].
__device__ inline v16bf ld_frag(const bf16* p) {
  Frag256 w;
  w.a = *(const uint4*)(p);
  w.b = *(const uint4*)(p + 8);
  return __builtin_bit_cast(v16bf, w);
}

__device__ inline v8f wmma_bf16(v16bf a, v16bf b, v8f c) {
  return __builtin_amdgcn_wmma_f32_16x16x32_bf16(
      false, a, false, b, (short)0, c, false, false);
}

#if USE_ASYNC
// LDS byte offset of a generic pointer that points into LDS.
__device__ inline unsigned lds_off(const void* p) {
  return (unsigned)(unsigned long long)
      (const __attribute__((address_space(3))) void*)(p);
}
// CDNA5 async global->LDS copy, 16 bytes per lane, tracked by ASYNCcnt.
__device__ inline void async_copy16(unsigned lds, const bf16* g) {
  asm volatile("global_load_async_to_lds_b128 %0, %1, off"
               :: "v"(lds), "v"(g) : "memory");
}
#define ASYNC_WAIT() asm volatile("s_wait_asynccnt 0x0" ::: "memory")
#else
#define ASYNC_WAIT()
#endif

// ---------------------------------------------------------------------------
// bf16 WMMA GEMM: C[M,N] = epi(A[M,K] * B[K,N] * scale + bias)
// A: bf16 row-major (lda). Bt: bf16 [N,K] row-major (pre-transposed, ldb=K).
// EPI: 0 none, 1 GELU(exact), 2 fusion sigmoid blend (aux0,aux1), 3 +aux0.
// Block 128 thr (4 waves), tile 64x64, K-step 64, double-buffered LDS,
// async global->LDS staging overlapped with WMMA.
// Requires: N % 64 == 0, K % 64 == 0, lda/ldb % 8 == 0.
// ---------------------------------------------------------------------------
template <int EPI>
__global__ __launch_bounds__(128) void gemm_k(
    const bf16* __restrict__ A, int lda,
    const bf16* __restrict__ Bt, int ldb,
    const float* __restrict__ bias,
    float* __restrict__ Cf, bf16* __restrict__ Cb, int ldc,
    int M, int N, int Kd, float scale,
    const float* __restrict__ aux0, const float* __restrict__ aux1) {
  __shared__ bf16 sA[2][64 * 64] __attribute__((aligned(32)));
  __shared__ bf16 sB[2][64 * 64] __attribute__((aligned(32)));
  const int tid  = threadIdx.x;
  const int lane = tid & 31, wave = tid >> 5;
  const int lo = lane & 15, hi = lane >> 4;
  const int wy = wave >> 1, wx = wave & 1;
  const int m0 = blockIdx.y * 64, n0 = blockIdx.x * 64;

  v8f acc[2][2];
#pragma unroll
  for (int i = 0; i < 2; i++)
#pragma unroll
    for (int j = 0; j < 2; j++) acc[i][j] = v8f_zero();

  // Per-thread staging coords: 4 chunks of 16B per tile (512 chunks total).
  int rowc[4], c8c[4], garow[4], gbrow[4];
#pragma unroll
  for (int i = 0; i < 4; i++) {
    int idx = tid + i * 128;
    rowc[i] = idx >> 3;            // 0..63
    c8c[i]  = idx & 7;             // 8B-bf16 chunk within row
    int ga = m0 + rowc[i]; garow[i] = (ga > M - 1) ? M - 1 : ga;
    int gb = n0 + rowc[i]; gbrow[i] = (gb > N - 1) ? N - 1 : gb;
  }

  const int nk = Kd >> 6;

  // -------- prologue: stage tile 0 --------
#if USE_ASYNC
#pragma unroll
  for (int i = 0; i < 4; i++) {
    async_copy16(lds_off(&sA[0][rowc[i] * 64 + c8c[i] * 8]),
                 A + (size_t)garow[i] * lda + c8c[i] * 8);
    async_copy16(lds_off(&sB[0][rowc[i] * 64 + c8c[i] * 8]),
                 Bt + (size_t)gbrow[i] * ldb + c8c[i] * 8);
  }
  ASYNC_WAIT();
  __syncthreads();
#else
  {
    uint4 ra[4], rb[4];
#pragma unroll
    for (int i = 0; i < 4; i++) {
      ra[i] = *(const uint4*)(A + (size_t)garow[i] * lda + c8c[i] * 8);
      rb[i] = *(const uint4*)(Bt + (size_t)gbrow[i] * ldb + c8c[i] * 8);
    }
#pragma unroll
    for (int i = 0; i < 4; i++) {
      *(uint4*)(&sA[0][rowc[i] * 64 + c8c[i] * 8]) = ra[i];
      *(uint4*)(&sB[0][rowc[i] * 64 + c8c[i] * 8]) = rb[i];
    }
  }
  __syncthreads();
#endif

  for (int t = 0; t < nk; t++) {
    const int cur = t & 1, nxt = cur ^ 1;
    const bool havenext = (t + 1 < nk);
    const int k1 = (t + 1) << 6;
#if USE_ASYNC
    if (havenext) {
#pragma unroll
      for (int i = 0; i < 4; i++) {
        async_copy16(lds_off(&sA[nxt][rowc[i] * 64 + c8c[i] * 8]),
                     A + (size_t)garow[i] * lda + k1 + c8c[i] * 8);
        async_copy16(lds_off(&sB[nxt][rowc[i] * 64 + c8c[i] * 8]),
                     Bt + (size_t)gbrow[i] * ldb + k1 + c8c[i] * 8);
      }
    }
#else
    uint4 ra[4], rb[4];
    if (havenext) {
#pragma unroll
      for (int i = 0; i < 4; i++) {
        ra[i] = *(const uint4*)(A + (size_t)garow[i] * lda + k1 + c8c[i] * 8);
        rb[i] = *(const uint4*)(Bt + (size_t)gbrow[i] * ldb + k1 + c8c[i] * 8);
      }
    }
#endif
    // -------- compute on buffer `cur`: 8 WMMAs --------
#pragma unroll
    for (int kk = 0; kk < 2; kk++) {
      v16bf af[2], bfr[2];
#pragma unroll
      for (int im = 0; im < 2; im++)
        af[im] = ld_frag(&sA[cur][(wy * 32 + im * 16 + lo) * 64 + kk * 32 + hi * 16]);
#pragma unroll
      for (int in = 0; in < 2; in++)
        bfr[in] = ld_frag(&sB[cur][(wx * 32 + in * 16 + lo) * 64 + kk * 32 + hi * 16]);
#pragma unroll
      for (int im = 0; im < 2; im++)
#pragma unroll
        for (int in = 0; in < 2; in++)
          acc[im][in] = wmma_bf16(af[im], bfr[in], acc[im][in]);
    }
#if USE_ASYNC
    ASYNC_WAIT();
    __syncthreads();
#else
    __syncthreads();
    if (havenext) {
#pragma unroll
      for (int i = 0; i < 4; i++) {
        *(uint4*)(&sA[nxt][rowc[i] * 64 + c8c[i] * 8]) = ra[i];
        *(uint4*)(&sB[nxt][rowc[i] * 64 + c8c[i] * 8]) = rb[i];
      }
    }
    __syncthreads();
#endif
  }

  // Epilogue. C layout: col = lane&15, row = vgpr + 8*(lane>>4).
#pragma unroll
  for (int im = 0; im < 2; im++) {
#pragma unroll
    for (int in = 0; in < 2; in++) {
      int col = n0 + wx * 32 + in * 16 + lo;
      float bv = bias ? bias[col] : 0.0f;
#pragma unroll
      for (int r = 0; r < 8; r++) {
        int row = m0 + wy * 32 + im * 16 + r + 8 * hi;
        if (row < M) {
          float v = acc[im][in][r] * scale + bv;
          size_t idx = (size_t)row * ldc + col;
          if (EPI == 1) {
            v = 0.5f * v * (1.0f + erff(v * 0.70710678118654752f));
          } else if (EPI == 2) {
            float g = 1.0f / (1.0f + expf(-v));
            v = g * aux0[idx] + (1.0f - g) * aux1[idx];
          } else if (EPI == 3) {
            v += aux0[idx];
          }
          if (Cf) Cf[idx] = v;
          if (Cb) Cb[idx] = f2bf(v);
        }
      }
    }
  }
}

// ---------------------------------------------------------------------------
// Flash attention (online softmax), head dim HD in {64, 256}.
// Grid: (ceil(sq/64), B*nheads). 128 threads = 4 waves; wave w owns 16 q-rows.
// ---------------------------------------------------------------------------
template <int HD>
__global__ __launch_bounds__(128) void flash_k(
    const bf16* __restrict__ Q, long q_bstride, int q_rstride,
    const bf16* __restrict__ Kp, const bf16* __restrict__ Vp,
    long kv_bstride, int kv_rstride,
    float* __restrict__ O, long o_bstride, int o_rstride,
    int nheads, int sq, int skv, float scale) {
  __shared__ bf16 sQ[64 * HD]  __attribute__((aligned(32)));
  __shared__ bf16 sK[64 * HD]  __attribute__((aligned(32)));
  __shared__ bf16 sVT[HD * 64] __attribute__((aligned(32)));
  __shared__ bf16 sP[4][16 * 64] __attribute__((aligned(32)));

  const int tid  = threadIdx.x;
  const int lane = tid & 31, wave = tid >> 5;
  const int lo = lane & 15, hi = lane >> 4;
  const int bh = blockIdx.y;
  const int b = bh / nheads, h = bh % nheads;
  const bf16* Qb = Q  + (size_t)b * q_bstride  + h * HD;
  const bf16* Kb = Kp + (size_t)b * kv_bstride + h * HD;
  const bf16* Vb = Vp + (size_t)b * kv_bstride + h * HD;
  float*      Ob = O  + (size_t)b * o_bstride  + h * HD;
  const int q0 = blockIdx.x * 64;
  constexpr int CH = 64 * HD / 8 / 128;  // 16B chunks per thread per tile

  // Stage Q tile (row-clamped so padding rows stay finite)
#if USE_ASYNC
#pragma unroll
  for (int i = 0; i < CH; i++) {
    int idx = tid + i * 128;
    int row = idx / (HD / 8), c8 = idx % (HD / 8);
    int qr = q0 + row; if (qr > sq - 1) qr = sq - 1;
    async_copy16(lds_off(&sQ[row * HD + c8 * 8]),
                 Qb + (size_t)qr * q_rstride + c8 * 8);
  }
#else
  {
    uint4 r[CH];
#pragma unroll
    for (int i = 0; i < CH; i++) {
      int idx = tid + i * 128;
      int row = idx / (HD / 8), c8 = idx % (HD / 8);
      int qr = q0 + row; if (qr > sq - 1) qr = sq - 1;
      r[i] = *(const uint4*)(Qb + (size_t)qr * q_rstride + c8 * 8);
    }
#pragma unroll
    for (int i = 0; i < CH; i++) {
      int idx = tid + i * 128;
      int row = idx / (HD / 8), c8 = idx % (HD / 8);
      *(uint4*)(&sQ[row * HD + c8 * 8]) = r[i];
    }
  }
#endif

  constexpr int ND = HD / 16;
  v8f Oacc[ND];
#pragma unroll
  for (int i = 0; i < ND; i++) Oacc[i] = v8f_zero();
  float mrow[8], lrow[8];
#pragma unroll
  for (int r = 0; r < 8; r++) { mrow[r] = -1e30f; lrow[r] = 0.0f; }
  bf16* sPw = sP[wave];

  for (int k0 = 0; k0 < skv; k0 += 64) {
    __syncthreads();   // previous compute done before restaging sK/sVT
    // Stage K tile [key][d]
#if USE_ASYNC
#pragma unroll
    for (int i = 0; i < CH; i++) {
      int idx = tid + i * 128;
      int row = idx / (HD / 8), c8 = idx % (HD / 8);
      async_copy16(lds_off(&sK[row * HD + c8 * 8]),
                   Kb + (size_t)(k0 + row) * kv_rstride + c8 * 8);
    }
#else
    {
      uint4 r[CH];
#pragma unroll
      for (int i = 0; i < CH; i++) {
        int idx = tid + i * 128;
        int row = idx / (HD / 8), c8 = idx % (HD / 8);
        r[i] = *(const uint4*)(Kb + (size_t)(k0 + row) * kv_rstride + c8 * 8);
      }
#pragma unroll
      for (int i = 0; i < CH; i++) {
        int idx = tid + i * 128;
        int row = idx / (HD / 8), c8 = idx % (HD / 8);
        *(uint4*)(&sK[row * HD + c8 * 8]) = r[i];
      }
    }
#endif
    // Stage V transposed [d][key] (register-batched; 16 dwords per batch)
    {
      constexpr int VCH = 64 * HD / 2 / 128;  // dword pairs per thread
#pragma unroll
      for (int b0 = 0; b0 < VCH; b0 += 16) {
        unsigned vr[16];
        int keys[16], dps[16];
#pragma unroll
        for (int j = 0; j < 16 && (b0 + j) < VCH; j++) {
          int idx = tid + (b0 + j) * 128;
          keys[j] = idx / (HD / 2); dps[j] = idx % (HD / 2);
          vr[j] = *(const unsigned*)(Vb + (size_t)(k0 + keys[j]) * kv_rstride + 2 * dps[j]);
        }
#pragma unroll
        for (int j = 0; j < 16 && (b0 + j) < VCH; j++) {
          sVT[(2 * dps[j]) * 64 + keys[j]]     = __builtin_bit_cast(bf16, (unsigned short)(vr[j] & 0xffffu));
          sVT[(2 * dps[j] + 1) * 64 + keys[j]] = __builtin_bit_cast(bf16, (unsigned short)(vr[j] >> 16));
        }
      }
    }
    ASYNC_WAIT();
    __syncthreads();

    // S = Q * K^T for this wave's 16 rows x 64 keys
    v8f sv[4];
#pragma unroll
    for (int jn = 0; jn < 4; jn++) {
      v8f s = v8f_zero();
#pragma unroll
      for (int kk = 0; kk < HD / 32; kk++) {
        v16bf a  = ld_frag(&sQ[(wave * 16 + lo) * HD + hi * 16 + kk * 32]);
        v16bf bb = ld_frag(&sK[(jn * 16 + lo)   * HD + hi * 16 + kk * 32]);
        s = wmma_bf16(a, bb, s);
      }
      sv[jn] = s;
    }

    // Online softmax (row = r + 8*hi; cols across 16-lane groups)
#pragma unroll
    for (int r = 0; r < 8; r++) {
      float mx = -1e30f;
#pragma unroll
      for (int jn = 0; jn < 4; jn++) { sv[jn][r] *= scale; mx = fmaxf(mx, sv[jn][r]); }
      mx = fmaxf(mx, __shfl_xor(mx, 1, 32));
      mx = fmaxf(mx, __shfl_xor(mx, 2, 32));
      mx = fmaxf(mx, __shfl_xor(mx, 4, 32));
      mx = fmaxf(mx, __shfl_xor(mx, 8, 32));
      float mnew = fmaxf(mrow[r], mx);
      float ef = expf(mrow[r] - mnew);
      float psum = 0.0f;
#pragma unroll
      for (int jn = 0; jn < 4; jn++) {
        float p = expf(sv[jn][r] - mnew);
        psum += p;
        sPw[(r + 8 * hi) * 64 + jn * 16 + lo] = f2bf(p);  // C-layout -> A-layout
      }
      psum += __shfl_xor(psum, 1, 32);
      psum += __shfl_xor(psum, 2, 32);
      psum += __shfl_xor(psum, 4, 32);
      psum += __shfl_xor(psum, 8, 32);
      lrow[r] = lrow[r] * ef + psum;
      mrow[r] = mnew;
#pragma unroll
      for (int jd = 0; jd < ND; jd++) Oacc[jd][r] *= ef;
    }

    // O += P * V  (K-dim = 64 keys -> 2 WMMA steps)
#pragma unroll
    for (int jd = 0; jd < ND; jd++) {
#pragma unroll
      for (int kk = 0; kk < 2; kk++) {
        v16bf a  = ld_frag(&sPw[lo * 64 + hi * 16 + kk * 32]);
        v16bf bb = ld_frag(&sVT[(jd * 16 + lo) * 64 + hi * 16 + kk * 32]);
        Oacc[jd] = wmma_bf16(a, bb, Oacc[jd]);
      }
    }
  }

  // Normalize and store
#pragma unroll
  for (int r = 0; r < 8; r++) {
    int row = q0 + wave * 16 + r + 8 * hi;
    if (row < sq) {
      float inv = 1.0f / lrow[r];
#pragma unroll
      for (int jd = 0; jd < ND; jd++)
        Ob[(size_t)row * o_rstride + jd * 16 + lo] = Oacc[jd][r] * inv;
    }
  }
}

// ---------------------------------------------------------------------------
// Small kernels
// ---------------------------------------------------------------------------
__global__ __launch_bounds__(256) void cvt_k(const float* __restrict__ src,
                                             bf16* __restrict__ dst,
                                             int sld, int dld, int Mr, int Nc) {
  long i = (long)blockIdx.x * 256 + threadIdx.x;
  long tot = (long)Mr * Nc;
  if (i < tot) {
    int r = (int)(i / Nc), c = (int)(i % Nc);
    dst[(size_t)r * dld + c] = f2bf(src[(size_t)r * sld + c]);
  }
}

// Weight transpose+convert: src f32 [K,N] row-major -> dst bf16 [N,K]
__global__ __launch_bounds__(256) void wt_k(const float* __restrict__ src,
                                            bf16* __restrict__ dst,
                                            int Kd, int N) {
  long i = (long)blockIdx.x * 256 + threadIdx.x;
  if (i < (long)Kd * N) {
    int n = (int)(i / Kd), k = (int)(i % Kd);
    dst[i] = f2bf(src[(size_t)k * N + n]);
  }
}

__global__ __launch_bounds__(256) void cid_k(const float* __restrict__ cache,
                                             const float* __restrict__ lid,
                                             bf16* __restrict__ out) {
  int i = blockIdx.x * 256 + threadIdx.x;
  const int tot = 4 * 768 * 320;
  if (i < tot) {
    int c = i % 320; int bt = i / 320; int t = bt % 768;
    float v = (c < 256) ? cache[(size_t)bt * 256 + c] : lid[(size_t)t * 64 + (c - 256)];
    out[i] = f2bf(v);
  }
}

// bias2[n] = Wd_b[n] + sum_l layer_embed[l] * Wd[256+l][n]
__global__ __launch_bounds__(256) void decbias_k(const float* __restrict__ Wd,
                                                 const float* __restrict__ bb,
                                                 const float* __restrict__ le,
                                                 float* __restrict__ out) {
  int n = blockIdx.x * 256 + threadIdx.x;
  if (n < 768) {
    float s = bb[n];
    for (int l = 0; l < 64; l++) s += le[l] * Wd[(size_t)(256 + l) * 768 + n];
    out[n] = s;
  }
}

__global__ __launch_bounds__(256) void ln_k(const float* __restrict__ src,
                                            const float* __restrict__ g,
                                            const float* __restrict__ b,
                                            float* __restrict__ outf,
                                            bf16* __restrict__ outb, int N) {
  __shared__ float red[256];
  int row = blockIdx.x, tid = threadIdx.x;
  const float* x = src + (size_t)row * N;
  float s = 0;
  for (int i = tid; i < N; i += 256) s += x[i];
  red[tid] = s; __syncthreads();
  for (int st = 128; st > 0; st >>= 1) { if (tid < st) red[tid] += red[tid + st]; __syncthreads(); }
  float mean = red[0] / N; __syncthreads();
  float v = 0;
  for (int i = tid; i < N; i += 256) { float d = x[i] - mean; v += d * d; }
  red[tid] = v; __syncthreads();
  for (int st = 128; st > 0; st >>= 1) { if (tid < st) red[tid] += red[tid + st]; __syncthreads(); }
  float rstd = rsqrtf(red[0] / N + 1e-5f);
  for (int i = tid; i < N; i += 256) {
    float o = (x[i] - mean) * rstd * g[i] + b[i];
    if (outf) outf[(size_t)row * N + i] = o;
    if (outb) outb[(size_t)row * N + i] = f2bf(o);
  }
}

__global__ __launch_bounds__(128) void rowdot_k(const float* __restrict__ X,
                                                const float* __restrict__ w,
                                                const float* __restrict__ wb,
                                                float* __restrict__ o1,
                                                float* __restrict__ o2,
                                                int Mr, int Kd) {
  int row = blockIdx.x * 4 + (threadIdx.x >> 5);
  int lane = threadIdx.x & 31;
  if (row >= Mr) return;
  const float* x = X + (size_t)row * Kd;
  float s = 0;
  for (int i = lane; i < Kd; i += 32) s += x[i] * w[i];
  for (int m = 16; m > 0; m >>= 1) s += __shfl_xor(s, m, 32);
  if (lane == 0) {
    float v = 1.0f / (1.0f + expf(-(s + wb[0])));
    o1[row] = v;
    if (o2) o2[row] = v;
  }
}

__global__ __launch_bounds__(256) void combine_k(const float* __restrict__ al,
                                                 const float* __restrict__ le,
                                                 const float* __restrict__ co,
                                                 float* __restrict__ out) {
  int i = blockIdx.x * 256 + threadIdx.x;
  if (i < 64 * 64) {
    float a = al[i >> 6];
    out[i] = a * le[i] + (1.0f - a) * co[i];
  }
}

__device__ inline float hash_u01(unsigned i) {
  unsigned h = i * 2654435761u + 0x9e3779b9u;
  h ^= h >> 16; h *= 0x85ebca6bu; h ^= h >> 13; h *= 0xc2b2ae35u; h ^= h >> 16;
  return (float)(h >> 8) * (1.0f / 16777216.0f);
}

__global__ __launch_bounds__(128) void slotsm_k(const float* __restrict__ comb,
                                                float* __restrict__ slotp,
                                                float* __restrict__ softp,
                                                float* __restrict__ rowent) {
  int row = blockIdx.x * 4 + (threadIdx.x >> 5);
  int lane = threadIdx.x & 31;
  float c0 = comb[row * 64 + lane], c1 = comb[row * 64 + lane + 32];
  float mx = fmaxf(c0, c1);
  for (int m = 16; m > 0; m >>= 1) mx = fmaxf(mx, __shfl_xor(mx, m, 32));
  float e0 = expf(c0 - mx), e1 = expf(c1 - mx);
  float s = e0 + e1;
  for (int m = 16; m > 0; m >>= 1) s += __shfl_xor(s, m, 32);
  float p0 = e0 / s, p1 = e1 / s;
  softp[row * 64 + lane] = p0; softp[row * 64 + lane + 32] = p1;
  float t = p0 * logf(p0 + 1e-8f) + p1 * logf(p1 + 1e-8f);
  for (int m = 16; m > 0; m >>= 1) t += __shfl_xor(t, m, 32);
  if (lane == 0) rowent[row] = -t;
  float u0 = hash_u01(row * 64 + lane), u1 = hash_u01(row * 64 + lane + 32);
  float g0 = -logf(-logf(u0 + 1e-8f) + 1e-8f);
  float g1 = -logf(-logf(u1 + 1e-8f) + 1e-8f);
  float a0 = c0 + g0, a1 = c1 + g1;
  mx = fmaxf(a0, a1);
  for (int m = 16; m > 0; m >>= 1) mx = fmaxf(mx, __shfl_xor(mx, m, 32));
  e0 = expf(a0 - mx); e1 = expf(a1 - mx);
  s = e0 + e1;
  for (int m = 16; m > 0; m >>= 1) s += __shfl_xor(s, m, 32);
  slotp[row * 64 + lane] = e0 / s; slotp[row * 64 + lane + 32] = e1 / s;
}

__global__ __launch_bounds__(64) void entred_k(const float* __restrict__ rowent,
                                               float* __restrict__ out) {
  __shared__ float red[64];
  int t = threadIdx.x;
  red[t] = rowent[t]; __syncthreads();
  for (int st = 32; st > 0; st >>= 1) { if (t < st) red[t] += red[t + st]; __syncthreads(); }
  if (t == 0) out[0] = red[0] / 64.0f;
}

__global__ __launch_bounds__(256) void counts_k(const float* __restrict__ slotp,
                                                float* __restrict__ out) {
  int i = blockIdx.x * 256 + threadIdx.x;
  if (i < 256) {
    int b = i >> 6, k = i & 63;
    float s = 0;
    for (int p = 0; p < 16; p++) s += slotp[(size_t)(b * 16 + p) * 64 + k];
    out[i] = s;
  }
}

// ---------------------------------------------------------------------------
// Host-side orchestration
// ---------------------------------------------------------------------------
extern "C" void kernel_launch(void* const* d_in, const int* in_sizes, int n_in,
                              void* d_out, int out_size, void* d_ws, size_t ws_size,
                              hipStream_t stream) {
  (void)in_sizes; (void)n_in; (void)out_size; (void)ws_size;
  const int B = 4, S = 2048, D = 768, H = 12, DC = 256, Ksl = 64, TS = 768, P = 16;
  const int BS = B * S;

  const float* x          = (const float*)d_in[0];
  const float* cache      = (const float*)d_in[1];
  const float* slot_emb   = (const float*)d_in[2];
  const float* layer_ids  = (const float*)d_in[3];
  const float* layer_emb  = (const float*)d_in[4];
  const float* saq_w = (const float*)d_in[5];  const float* saq_b = (const float*)d_in[6];
  const float* sak_w = (const float*)d_in[7];  const float* sak_b = (const float*)d_in[8];
  const float* sav_w = (const float*)d_in[9];  const float* sav_b = (const float*)d_in[10];
  const float* sao_w = (const float*)d_in[11]; const float* sao_b = (const float*)d_in[12];
  const float* paq_w = (const float*)d_in[13]; const float* paq_b = (const float*)d_in[14];
  const float* pak_w = (const float*)d_in[15]; const float* pak_b = (const float*)d_in[16];
  const float* pav_w = (const float*)d_in[17]; const float* pav_b = (const float*)d_in[18];
  const float* pao_w = (const float*)d_in[19]; const float* pao_b = (const float*)d_in[20];
  const float* ffn1_w = (const float*)d_in[21]; const float* ffn1_b = (const float*)d_in[22];
  const float* ffn2_w = (const float*)d_in[23]; const float* ffn2_b = (const float*)d_in[24];
  const float* n1_g = (const float*)d_in[25]; const float* n1_b = (const float*)d_in[26];
  const float* n2_g = (const float*)d_in[27]; const float* n2_b = (const float*)d_in[28];
  const float* patq = (const float*)d_in[29];
  const float* gate_w = (const float*)d_in[30]; const float* gate_b = (const float*)d_in[31];
  const float* sel_w = (const float*)d_in[32];  const float* sel_b = (const float*)d_in[33];
  const float* squ_w = (const float*)d_in[34];  const float* squ_b = (const float*)d_in[35];
  const float* sky_w = (const float*)d_in[36];  const float* sky_b = (const float*)d_in[37];
  const float* an_w = (const float*)d_in[38];   const float* an_b = (const float*)d_in[39];
  const float* wc_w = (const float*)d_in[40];   const float* wc_b = (const float*)d_in[41];
  const float* wd_w = (const float*)d_in[42];   const float* wd_b = (const float*)d_in[43];
  const float* cqr_w = (const float*)d_in[44];  const float* cqr_b = (const float*)d_in[45];
  const float* cak_w = (const float*)d_in[46];  const float* cak_b = (const float*)d_in[47];
  const float* cav_w = (const float*)d_in[48];  const float* cav_b = (const float*)d_in[49];
  const float* fg_w = (const float*)d_in[50];   const float* fg_b = (const float*)d_in[51];

  float* out_f        = (float*)d_out;
  float* y_out        = out_f;
  float* pcache_out   = out_f + 6291456;
  float* scores_out   = pcache_out + 16384;
  float* slotp_out    = scores_out + 64;
  float* softp_out    = slotp_out + 4096;
  float* counts_out   = softp_out + 4096;
  float* ent_out      = counts_out + 256;
  float* alpha_out    = ent_out + 1;
  float* patterns_out = alpha_out + 64;

  char* base = (char*)d_ws; size_t off = 0;
  auto alloc = [&](size_t bytes) -> void* {
    off = (off + 255) & ~(size_t)255;
    void* p = base + off; off += bytes; return p;
  };
  // transposed bf16 weights [N,K]
  bf16* saqT = (bf16*)alloc((size_t)768 * 768 * 2);
  bf16* sakT = (bf16*)alloc((size_t)768 * 768 * 2);
  bf16* savT = (bf16*)alloc((size_t)768 * 768 * 2);
  bf16* saoT = (bf16*)alloc((size_t)768 * 768 * 2);
  bf16* paqT = (bf16*)alloc((size_t)768 * 768 * 2);
  bf16* pakT = (bf16*)alloc((size_t)768 * 768 * 2);
  bf16* pavT = (bf16*)alloc((size_t)768 * 768 * 2);
  bf16* paoT = (bf16*)alloc((size_t)768 * 768 * 2);
  bf16* ffn1T = (bf16*)alloc((size_t)3072 * 768 * 2);
  bf16* ffn2T = (bf16*)alloc((size_t)768 * 3072 * 2);
  bf16* fgT   = (bf16*)alloc((size_t)768 * 1536 * 2);
  bf16* cqrT  = (bf16*)alloc((size_t)256 * 768 * 2);
  bf16* cakT  = (bf16*)alloc((size_t)256 * 320 * 2);
  bf16* cavT  = (bf16*)alloc((size_t)256 * 320 * 2);
  bf16* wdT   = (bf16*)alloc((size_t)768 * 256 * 2);
  bf16* selT  = (bf16*)alloc((size_t)64 * 768 * 2);
  bf16* squT  = (bf16*)alloc((size_t)256 * 768 * 2);
  bf16* skyT  = (bf16*)alloc((size_t)256 * 256 * 2);
  bf16* wcT   = (bf16*)alloc((size_t)256 * 768 * 2);
  // activations
  bf16*  x_bf      = (bf16*)alloc((size_t)BS * D * 2);
  bf16*  cid_bf    = (bf16*)alloc((size_t)B * TS * 320 * 2);
  bf16*  cq_bf     = (bf16*)alloc((size_t)BS * DC * 2);
  bf16*  ck_bf     = (bf16*)alloc((size_t)B * TS * DC * 2);
  bf16*  cv_bf     = (bf16*)alloc((size_t)B * TS * DC * 2);
  float* raw_f     = (float*)alloc((size_t)BS * DC * 4);
  bf16*  raw_bf    = (bf16*)alloc((size_t)BS * DC * 2);
  float* dbias     = (float*)alloc(768 * 4);
  float* context_f = (float*)alloc((size_t)BS * D * 4);   // reused by ffnh
  bf16*  cat_bf    = (bf16*)alloc((size_t)BS * 2 * D * 2);
  float* xf_f      = (float*)alloc((size_t)BS * D * 4);   // reused as pre-LN2
  bf16*  xf_bf     = (bf16*)alloc((size_t)BS * D * 2);
  bf16*  q_bf      = (bf16*)alloc((size_t)BS * D * 2);    // reused as pattern-K
  bf16*  k_bf      = (bf16*)alloc((size_t)BS * D * 2);    // reused as pattern-V
  bf16*  v_bf      = (bf16*)alloc((size_t)BS * D * 2);
  float* attn_f    = (float*)alloc((size_t)BS * D * 4);   // reused for attn+res
  bf16*  attn_bf   = (bf16*)alloc((size_t)BS * D * 2);
  float* xf1_f     = (float*)alloc((size_t)BS * D * 4);
  bf16*  xf1_bf    = (bf16*)alloc((size_t)BS * D * 2);
  bf16*  y_bf      = (bf16*)alloc((size_t)BS * D * 2);
  bf16*  pq_bf     = (bf16*)alloc((size_t)P * D * 2);
  bf16*  patq_bf   = (bf16*)alloc((size_t)P * D * 2);
  float* pato_f    = (float*)alloc((size_t)B * P * D * 4);
  bf16*  pato_bf   = (bf16*)alloc((size_t)B * P * D * 2);
  bf16*  patterns_bf = (bf16*)alloc((size_t)B * P * D * 2);
  bf16*  se_bf     = (bf16*)alloc((size_t)Ksl * DC * 2);
  bf16*  qs_bf     = (bf16*)alloc((size_t)B * P * DC * 2);
  bf16*  ks_bf     = (bf16*)alloc((size_t)Ksl * DC * 2);
  float* learned_f = (float*)alloc((size_t)B * P * Ksl * 4);
  float* content_f = (float*)alloc((size_t)B * P * Ksl * 4);
  float* combined_f = (float*)alloc((size_t)B * P * Ksl * 4);
  float* alpha_buf = (float*)alloc(64 * 4);
  float* rowent_f  = (float*)alloc(64 * 4);
  bf16*  ffnh_bf   = (bf16*)context_f;   // reuse 50.3MB region
  float* attnres_f = attn_f;
  float* preln2_f  = xf_f;
  bf16*  pk_bf = q_bf, * pv_bf = k_bf;

  auto cvt = [&](const float* s, bf16* dd, int sld, int dld, int Mr, int Nc) {
    long tot = (long)Mr * Nc;
    cvt_k<<<dim3((unsigned)((tot + 255) / 256)), dim3(256), 0, stream>>>(s, dd, sld, dld, Mr, Nc);
  };
  auto wt = [&](const float* s, bf16* dd, int Kd, int N) {
    long tot = (long)Kd * N;
    wt_k<<<dim3((unsigned)((tot + 255) / 256)), dim3(256), 0, stream>>>(s, dd, Kd, N);
  };
  auto gemm = [&](const bf16* A, int lda, const bf16* Bt, int ldb,
                  const float* bias, float* Cf, bf16* Cb, int ldc,
                  int M, int N, int Kd, float scale, int epi,
                  const float* a0, const float* a1) {
    dim3 g(N / 64, (M + 63) / 64), blk(128);
    switch (epi) {
      case 0: gemm_k<0><<<g, blk, 0, stream>>>(A, lda, Bt, ldb, bias, Cf, Cb, ldc, M, N, Kd, scale, a0, a1); break;
      case 1: gemm_k<1><<<g, blk, 0, stream>>>(A, lda, Bt, ldb, bias, Cf, Cb, ldc, M, N, Kd, scale, a0, a1); break;
      case 2: gemm_k<2><<<g, blk, 0, stream>>>(A, lda, Bt, ldb, bias, Cf, Cb, ldc, M, N, Kd, scale, a0, a1); break;
      default: gemm_k<3><<<g, blk, 0, stream>>>(A, lda, Bt, ldb, bias, Cf, Cb, ldc, M, N, Kd, scale, a0, a1); break;
    }
  };

  // ---- stage 0: weight transposes + activation conversions
  wt(saq_w, saqT, 768, 768);  wt(sak_w, sakT, 768, 768);
  wt(sav_w, savT, 768, 768);  wt(sao_w, saoT, 768, 768);
  wt(paq_w, paqT, 768, 768);  wt(pak_w, pakT, 768, 768);
  wt(pav_w, pavT, 768, 768);  wt(pao_w, paoT, 768, 768);
  wt(ffn1_w, ffn1T, 768, 3072);
  wt(ffn2_w, ffn2T, 3072, 768);
  wt(fg_w, fgT, 1536, 768);
  wt(cqr_w, cqrT, 768, 256);
  wt(cak_w, cakT, 320, 256);
  wt(cav_w, cavT, 320, 256);
  wt(wd_w, wdT, 256, 768);    // top 256 rows of W_decompress
  wt(sel_w, selT, 768, 64);
  wt(squ_w, squT, 768, 256);
  wt(sky_w, skyT, 256, 256);
  wt(wc_w, wcT, 768, 256);
  cvt(x, x_bf, D, D, BS, D);
  cid_k<<<dim3((4 * 768 * 320 + 255) / 256), dim3(256), 0, stream>>>(cache, layer_ids, cid_bf);
  cvt(patq, pq_bf, D, D, P, D);
  cvt(slot_emb, se_bf, DC, DC, Ksl, DC);

  // ---- stage 1: cache read attention (d=256, 768 KV slots)
  gemm(x_bf, D, cqrT, D, cqr_b, nullptr, cq_bf, DC, BS, DC, D, 1.f, 0, 0, 0);
  gemm(cid_bf, 320, cakT, 320, cak_b, nullptr, ck_bf, DC, B * TS, DC, 320, 1.f, 0, 0, 0);
  gemm(cid_bf, 320, cavT, 320, cav_b, nullptr, cv_bf, DC, B * TS, DC, 320, 1.f, 0, 0, 0);
  flash_k<256><<<dim3(S / 64, B), dim3(128), 0, stream>>>(
      cq_bf, (long)S * DC, DC, ck_bf, cv_bf, (long)TS * DC, DC,
      raw_f, (long)S * DC, DC, 1, S, TS, 0.0625f);
  cvt(raw_f, raw_bf, DC, DC, BS, DC);
  decbias_k<<<dim3(3), dim3(256), 0, stream>>>(wd_w, wd_b, layer_emb, dbias);
  gemm(raw_bf, DC, wdT, DC, dbias, context_f, nullptr, D, BS, D, DC, 1.f, 0, 0, 0);

  // ---- stage 2: gated fusion (K=1536 over concat)
  cvt(x, cat_bf, D, 2 * D, BS, D);
  cvt(context_f, cat_bf + D, D, 2 * D, BS, D);
  gemm(cat_bf, 2 * D, fgT, 2 * D, fg_b, xf_f, xf_bf, D, BS, D, 2 * D, 1.f, 2, x, context_f);

  // ---- stage 3: self attention
  gemm(xf_bf, D, saqT, D, saq_b, nullptr, q_bf, D, BS, D, D, 1.f, 0, 0, 0);
  gemm(xf_bf, D, sakT, D, sak_b, nullptr, k_bf, D, BS, D, D, 1.f, 0, 0, 0);
  gemm(xf_bf, D, savT, D, sav_b, nullptr, v_bf, D, BS, D, D, 1.f, 0, 0, 0);
  flash_k<64><<<dim3(S / 64, B * H), dim3(128), 0, stream>>>(
      q_bf, (long)S * D, D, k_bf, v_bf, (long)S * D, D,
      attn_f, (long)S * D, D, H, S, S, 0.125f);
  cvt(attn_f, attn_bf, D, D, BS, D);
  gemm(attn_bf, D, saoT, D, sao_b, attnres_f, nullptr, D, BS, D, D, 1.f, 3, xf_f, 0);
  ln_k<<<dim3(BS), dim3(256), 0, stream>>>(attnres_f, n1_g, n1_b, xf1_f, xf1_bf, D);

  // ---- stage 4: FFN with exact GELU
  gemm(xf1_bf, D, ffn1T, D, ffn1_b, nullptr, ffnh_bf, 4 * D, BS, 4 * D, D, 1.f, 1, 0, 0);
  gemm(ffnh_bf, 4 * D, ffn2T, 4 * D, ffn2_b, preln2_f, nullptr, D, BS, D, 4 * D, 1.f, 3, xf1_f, 0);
  ln_k<<<dim3(BS), dim3(256), 0, stream>>>(preln2_f, n2_g, n2_b, y_out, y_bf, D);

  // ---- stage 5: pattern attention
  gemm(pq_bf, D, paqT, D, paq_b, nullptr, patq_bf, D, P, D, D, 1.f, 0, 0, 0);
  gemm(y_bf, D, pakT, D, pak_b, nullptr, pk_bf, D, BS, D, D, 1.f, 0, 0, 0);
  gemm(y_bf, D, pavT, D, pav_b, nullptr, pv_bf, D, BS, D, D, 1.f, 0, 0, 0);
  flash_k<64><<<dim3(1, B * H), dim3(128), 0, stream>>>(
      patq_bf, 0L, D, pk_bf, pv_bf, (long)S * D, D,
      pato_f, (long)P * D, D, H, P, S, 0.125f);
  cvt(pato_f, pato_bf, D, D, B * P, D);
  gemm(pato_bf, D, paoT, D, pao_b, patterns_out, patterns_bf, D, B * P, D, D, 1.f, 0, 0, 0);

  // ---- stage 6: slot selection
  rowdot_k<<<dim3(16), dim3(128), 0, stream>>>(patterns_out, gate_w, gate_b, scores_out, nullptr, B * P, D);
  rowdot_k<<<dim3(16), dim3(128), 0, stream>>>(patterns_out, an_w, an_b, alpha_buf, alpha_out, B * P, D);
  gemm(patterns_bf, D, selT, D, sel_b, learned_f, nullptr, Ksl, B * P, Ksl, D, 1.f, 0, 0, 0);
  gemm(patterns_bf, D, squT, D, squ_b, nullptr, qs_bf, DC, B * P, DC, D, 1.f, 0, 0, 0);
  gemm(se_bf, DC, skyT, DC, sky_b, nullptr, ks_bf, DC, Ksl, DC, DC, 1.f, 0, 0, 0);
  gemm(qs_bf, DC, ks_bf, DC, nullptr, content_f, nullptr, Ksl, B * P, Ksl, DC, 0.0625f, 0, 0, 0);
  combine_k<<<dim3(16), dim3(256), 0, stream>>>(alpha_buf, learned_f, content_f, combined_f);
  slotsm_k<<<dim3(16), dim3(128), 0, stream>>>(combined_f, slotp_out, softp_out, rowent_f);
  entred_k<<<dim3(1), dim3(64), 0, stream>>>(rowent_f, ent_out);
  counts_k<<<dim3(1), dim3(256), 0, stream>>>(slotp_out, counts_out);
  gemm(patterns_bf, D, wcT, D, wc_b, pcache_out, nullptr, DC, B * P, DC, D, 1.f, 0, 0, 0);
}
